// NABlock_41661182771218
// MI455X (gfx1250) — matrix operand order repaired and checked by reading, not compiled, not run on hardware
//
#include <hip/hip_runtime.h>
#include <hip/hip_bf16.h>
#include <math.h>

typedef __attribute__((ext_vector_type(16))) _Float16 v16h;
typedef __attribute__((ext_vector_type(8)))  _Float16 v8h;
typedef __attribute__((ext_vector_type(8)))  float    v8f;

#define NB     2
#define CH     64
#define HH     128
#define WW     128
#define PIX    (HH*WW)       // 16384
#define MROWS  (NB*PIX)      // 32768
#define HEADS  2
#define DH     32
#define KSZ    7
#define C4     256

// ---------------------------------------------------------------------------
// f32 -> f16 weight conversion
// ---------------------------------------------------------------------------
__global__ void f32_to_f16_kernel(const float* __restrict__ src,
                                  _Float16* __restrict__ dst, int n) {
    int i = blockIdx.x * blockDim.x + threadIdx.x;
    if (i < n) dst[i] = (_Float16)src[i];
}

// ---------------------------------------------------------------------------
// NCHW -> token-major transpose fused with LayerNorm1, emits f16 activations.
// Block = 256 threads, handles a 64(ch) x 256(pixel) tile via LDS.
// ---------------------------------------------------------------------------
__global__ void ln1_kernel(const float* __restrict__ x,
                           const float* __restrict__ w,
                           const float* __restrict__ b,
                           _Float16* __restrict__ xn) {
    __shared__ float tile[CH][256];
    const int t = threadIdx.x;
    const int n = blockIdx.x >> 6;                // 64 blocks per image
    const int pixbase = (blockIdx.x & 63) << 8;
    const float* xp = x + (size_t)n * CH * PIX + pixbase + t;
    #pragma unroll
    for (int c = 0; c < CH; ++c) tile[c][t] = xp[(size_t)c * PIX];
    __syncthreads();

    float s = 0.f;
    #pragma unroll
    for (int c = 0; c < CH; ++c) s += tile[c][t];
    const float mean = s * (1.f / CH);
    float vs = 0.f;
    #pragma unroll
    for (int c = 0; c < CH; ++c) { float d = tile[c][t] - mean; vs += d * d; }
    const float rstd = rsqrtf(vs * (1.f / CH) + 1e-5f);

    const size_t row = (size_t)n * PIX + pixbase + t;
    _Float16* op = xn + row * CH;
    #pragma unroll
    for (int c = 0; c < CH; ++c)
        op[c] = (_Float16)((tile[c][t] - mean) * rstd * w[c] + b[c]);
}

// ---------------------------------------------------------------------------
// WMMA strip GEMM helpers.  A row-major (M x K f16, lda == K), W row-major
// (N x K f16) => D = A * W^T with f32 accumulate (v_wmma_f32_16x16x32_f16).
// Each wave owns a 16-row strip of A: A fragments are loaded into registers
// ONCE and reused across every 16-col N tile (A is the large operand; the
// weights are tiny and L2-resident, so this is the traffic that matters).
//
// A frag: lane<16 holds row M=lane, K k0..k0+15;
//         lane>=16 holds K k0+16..k0+31.
// B frag: lane<16 holds col N=lane, K k0..k0+15 (packed pairs);
//         lane>=16 holds K k0+16..k0+31 — 32 contiguous bytes of W's row.
// C/D:    row = row0 + (lane>=16 ? 8 : 0) + j ; col = col0 + (lane & 15)
// ---------------------------------------------------------------------------
template<int K>
__device__ inline void load_a_frags(const _Float16* __restrict__ A, int row0,
                                    v16h* __restrict__ af) {
    const int lane = threadIdx.x & 31;
    const int hi   = lane >> 4;
    const int l15  = lane & 15;
    const _Float16* ap = A + (size_t)(row0 + l15) * K + (hi << 4);
    #pragma unroll
    for (int kk = 0; kk < K / 32; ++kk) {
        v8h lo = *(const v8h*)(ap + kk * 32);
        v8h hh = *(const v8h*)(ap + kk * 32 + 8);
        af[kk] = __builtin_shufflevector(lo, hh,
                    0,1,2,3,4,5,6,7,8,9,10,11,12,13,14,15);
    }
}

template<int K>
__device__ inline v8f wmma_strip_tile(const v16h* __restrict__ af,
                                      const _Float16* __restrict__ W, int col0) {
    const int lane = threadIdx.x & 31;
    const int hi   = lane >> 4;
    const int l15  = lane & 15;
    const _Float16* bp = W + (size_t)(col0 + l15) * K + (hi << 4);
    v8f acc = {};
    #pragma unroll
    for (int kk = 0; kk < K / 32; ++kk) {
        v16h bf = *(const v16h*)(bp + kk * 32);
        acc = __builtin_amdgcn_wmma_f32_16x16x32_f16(
                  false, af[kk], false, bf, (short)0, acc, false, false);
    }
    return acc;
}

// ---------------------------------------------------------------------------
// QKV projection: (M x 64) @ (192 x 64)^T, fused bias + q-scale (dh^-0.5).
// One wave per 16-row strip, sweeps all 12 N tiles with A in registers.
// ---------------------------------------------------------------------------
__global__ void qkv_gemm_kernel(const _Float16* __restrict__ xn,
                                const _Float16* __restrict__ wq,
                                const float* __restrict__ qb,
                                float* __restrict__ qkv) {
    const int strip = blockIdx.x * 8 + (threadIdx.x >> 5);
    const int row0  = strip * 16;
    v16h af[CH / 32];
    load_a_frags<CH>(xn, row0, af);

    const int lane  = threadIdx.x & 31;
    const int l15   = lane & 15;
    const int rbase = row0 + ((lane >> 4) << 3);
    #pragma unroll
    for (int nt = 0; nt < 192 / 16; ++nt) {
        v8f acc = wmma_strip_tile<CH>(af, wq, nt * 16);
        const int col = nt * 16 + l15;
        const float bias  = qb[col];
        const float scale = (col < CH) ? 0.17677669529663689f : 1.0f; // 1/sqrt(32)
        #pragma unroll
        for (int j = 0; j < 8; ++j)
            qkv[(size_t)(rbase + j) * 192 + col] = (acc[j] + bias) * scale;
    }
}

// ---------------------------------------------------------------------------
// Neighborhood attention: one wave32 per (token, head); lane = dh element.
// Online softmax; scores via wave shuffle-tree dot products; rpb bias.
// ---------------------------------------------------------------------------
__global__ void attn_kernel(const float* __restrict__ qkv,
                            const float* __restrict__ rpb,
                            _Float16* __restrict__ ctx) {
    const int id   = blockIdx.x * 8 + (threadIdx.x >> 5);
    const int lane = threadIdx.x & 31;
    const int head = id & 1;
    const int p    = id >> 1;              // token index
    const int n    = p >> 14;
    const int hw   = p & (PIX - 1);
    const int h    = hw >> 7, w = hw & (WW - 1);

    const float q = qkv[(size_t)p * 192 + head * DH + lane];
    int sh = h - (KSZ / 2); sh = sh < 0 ? 0 : (sh > HH - KSZ ? HH - KSZ : sh);
    int sw = w - (KSZ / 2); sw = sw < 0 ? 0 : (sw > WW - KSZ ? WW - KSZ : sw);
    const float* rp = rpb + head * 169;    // 13x13 table per head

    float m = -INFINITY, l = 0.f, o = 0.f;
    for (int i = 0; i < KSZ; ++i) {
        const int kh = sh + i;
        const int relh = h - kh + (KSZ - 1);
        for (int j = 0; j < KSZ; ++j) {
            const int kw = sw + j;
            const size_t kr = (size_t)n * PIX + kh * WW + kw;
            const float* kp = qkv + kr * 192 + 64 + head * DH;
            float part = q * kp[lane];
            #pragma unroll
            for (int off = 16; off > 0; off >>= 1)
                part += __shfl_xor(part, off, 32);
            const float s = part + rp[relh * 13 + (w - kw + (KSZ - 1))];
            const float mn   = fmaxf(m, s);
            const float corr = __expf(m - mn);
            const float pw   = __expf(s - mn);
            l = l * corr + pw;
            o = o * corr + pw * kp[64 + lane];   // v is 64 floats past k
            m = mn;
        }
    }
    ctx[(size_t)p * CH + head * DH + lane] = (_Float16)(o / l);
}

// ---------------------------------------------------------------------------
// Output projection + residual: x2 = transpose(x) + ctx @ proj_w^T + b
// ---------------------------------------------------------------------------
__global__ void proj_gemm_kernel(const _Float16* __restrict__ ctx,
                                 const _Float16* __restrict__ wp,
                                 const float* __restrict__ pb,
                                 const float* __restrict__ x,
                                 float* __restrict__ x2) {
    const int strip = blockIdx.x * 8 + (threadIdx.x >> 5);
    const int row0  = strip * 16;
    v16h af[CH / 32];
    load_a_frags<CH>(ctx, row0, af);

    const int lane  = threadIdx.x & 31;
    const int l15   = lane & 15;
    const int rbase = row0 + ((lane >> 4) << 3);
    #pragma unroll
    for (int nt = 0; nt < CH / 16; ++nt) {
        v8f acc = wmma_strip_tile<CH>(af, wp, nt * 16);
        const int col  = nt * 16 + l15;
        const float bias = pb[col];
        #pragma unroll
        for (int j = 0; j < 8; ++j) {
            const int row = rbase + j;
            const int n = row >> 14, pix = row & (PIX - 1);
            const float sc = x[(size_t)n * CH * PIX + (size_t)col * PIX + pix];
            x2[(size_t)row * CH + col] = sc + acc[j] + bias;
        }
    }
}

// ---------------------------------------------------------------------------
// LayerNorm2: one wave per token (64 channels = 2 per lane), shuffle reductions
// ---------------------------------------------------------------------------
__global__ void ln2_kernel(const float* __restrict__ x2,
                           const float* __restrict__ w,
                           const float* __restrict__ b,
                           _Float16* __restrict__ y) {
    const int p = blockIdx.x * 8 + (threadIdx.x >> 5);
    const int lane = threadIdx.x & 31;
    const float v0 = x2[(size_t)p * CH + lane];
    const float v1 = x2[(size_t)p * CH + 32 + lane];
    float s = v0 + v1;
    #pragma unroll
    for (int off = 16; off > 0; off >>= 1) s += __shfl_xor(s, off, 32);
    const float mean = s * (1.f / CH);
    const float d0 = v0 - mean, d1 = v1 - mean;
    float vs = d0 * d0 + d1 * d1;
    #pragma unroll
    for (int off = 16; off > 0; off >>= 1) vs += __shfl_xor(vs, off, 32);
    const float rstd = rsqrtf(vs * (1.f / CH) + 1e-5f);
    y[(size_t)p * CH + lane]      = (_Float16)(d0 * rstd * w[lane] + b[lane]);
    y[(size_t)p * CH + 32 + lane] = (_Float16)(d1 * rstd * w[32 + lane] + b[32 + lane]);
}

// ---------------------------------------------------------------------------
// FC1: (M x 64) @ (256 x 64)^T, fused bias + exact GELU, f16 out
// ---------------------------------------------------------------------------
__global__ void fc1_gemm_kernel(const _Float16* __restrict__ y,
                                const _Float16* __restrict__ w1,
                                const float* __restrict__ b1,
                                _Float16* __restrict__ hbuf) {
    const int strip = blockIdx.x * 8 + (threadIdx.x >> 5);
    const int row0  = strip * 16;
    v16h af[CH / 32];
    load_a_frags<CH>(y, row0, af);

    const int lane  = threadIdx.x & 31;
    const int l15   = lane & 15;
    const int rbase = row0 + ((lane >> 4) << 3);
    #pragma unroll
    for (int nt = 0; nt < C4 / 16; ++nt) {
        v8f acc = wmma_strip_tile<CH>(af, w1, nt * 16);
        const int col  = nt * 16 + l15;
        const float bias = b1[col];
        #pragma unroll
        for (int j = 0; j < 8; ++j) {
            const float v = acc[j] + bias;
            const float g = 0.5f * v * (1.f + erff(v * 0.7071067811865475f));
            hbuf[(size_t)(rbase + j) * C4 + col] = (_Float16)g;
        }
    }
}

// ---------------------------------------------------------------------------
// FC2: (M x 256) @ (64 x 256)^T + bias + residual, fused token->NCHW store
// ---------------------------------------------------------------------------
__global__ void fc2_gemm_kernel(const _Float16* __restrict__ hbuf,
                                const _Float16* __restrict__ w2,
                                const float* __restrict__ b2,
                                const float* __restrict__ x2,
                                float* __restrict__ out) {
    const int strip = blockIdx.x * 8 + (threadIdx.x >> 5);
    const int row0  = strip * 16;
    v16h af[C4 / 32];                       // 8 frags, 64 VGPRs: K=256 in regs
    load_a_frags<C4>(hbuf, row0, af);

    const int lane  = threadIdx.x & 31;
    const int l15   = lane & 15;
    const int rbase = row0 + ((lane >> 4) << 3);
    #pragma unroll
    for (int nt = 0; nt < CH / 16; ++nt) {
        v8f acc = wmma_strip_tile<C4>(af, w2, nt * 16);
        const int col  = nt * 16 + l15;
        const float bias = b2[col];
        #pragma unroll
        for (int j = 0; j < 8; ++j) {
            const int row = rbase + j;
            const int n = row >> 14, pix = row & (PIX - 1);
            const float v = x2[(size_t)row * CH + col] + acc[j] + bias;
            out[(size_t)n * CH * PIX + (size_t)col * PIX + pix] = v;
        }
    }
}

// ---------------------------------------------------------------------------
// Host-side orchestration
// ---------------------------------------------------------------------------
extern "C" void kernel_launch(void* const* d_in, const int* in_sizes, int n_in,
                              void* d_out, int out_size, void* d_ws, size_t ws_size,
                              hipStream_t stream) {
    const float* x      = (const float*)d_in[0];
    const float* qkv_w  = (const float*)d_in[1];
    const float* qkv_b  = (const float*)d_in[2];
    const float* proj_w = (const float*)d_in[3];
    const float* proj_b = (const float*)d_in[4];
    const float* rpb    = (const float*)d_in[5];
    const float* ln1_w  = (const float*)d_in[6];
    const float* ln1_b  = (const float*)d_in[7];
    const float* ln2_w  = (const float*)d_in[8];
    const float* ln2_b  = (const float*)d_in[9];
    const float* fc1_w  = (const float*)d_in[10];
    const float* fc1_b  = (const float*)d_in[11];
    const float* fc2_w  = (const float*)d_in[12];
    const float* fc2_b  = (const float*)d_in[13];
    float* out = (float*)d_out;

    // Workspace layout (40.1 MB total, aggressive reuse):
    //   [0, 24MB)        qkv f32 (M x 192)   -> reused as hbuf f16 (M x 256)
    //   [24MB, 32MB)     x2  f32 (M x 64)
    //   [32MB, 36MB)     xn  f16 (M x 64)    -> reused as y (LN2 out)
    //   [36MB, 40MB)     ctx f16 (M x 64)
    //   [40MB, +96KB)    f16 weight copies
    char* ws = (char*)d_ws;
    float*    qkv   = (float*)(ws);
    float*    x2    = (float*)(ws + 25165824);
    _Float16* xn    = (_Float16*)(ws + 25165824 + 8388608);
    _Float16* ctx   = (_Float16*)(ws + 25165824 + 8388608 + 4194304);
    char*     wbase = ws + 25165824 + 8388608 + 4194304 + 4194304;
    _Float16* wqkv  = (_Float16*)(wbase);
    _Float16* wproj = (_Float16*)(wbase + 24576);
    _Float16* wfc1  = (_Float16*)(wbase + 24576 + 8192);
    _Float16* wfc2  = (_Float16*)(wbase + 24576 + 8192 + 32768);
    _Float16* hbuf  = (_Float16*)(ws);   // aliases qkv (dead by then)

    f32_to_f16_kernel<<<(3 * CH * CH + 255) / 256, 256, 0, stream>>>(qkv_w, wqkv, 3 * CH * CH);
    f32_to_f16_kernel<<<(CH * CH + 255) / 256, 256, 0, stream>>>(proj_w, wproj, CH * CH);
    f32_to_f16_kernel<<<(C4 * CH + 255) / 256, 256, 0, stream>>>(fc1_w, wfc1, C4 * CH);
    f32_to_f16_kernel<<<(CH * C4 + 255) / 256, 256, 0, stream>>>(fc2_w, wfc2, CH * C4);

    ln1_kernel<<<MROWS / 256, 256, 0, stream>>>(x, ln1_w, ln1_b, xn);

    // Strip GEMMs: one wave per 16-row strip, 8 waves per block.
    qkv_gemm_kernel<<<MROWS / 16 / 8, 256, 0, stream>>>(xn, wqkv, qkv_b, qkv);

    attn_kernel<<<(MROWS * HEADS) / 8, 256, 0, stream>>>(qkv, rpb, ctx);

    proj_gemm_kernel<<<MROWS / 16 / 8, 256, 0, stream>>>(ctx, wproj, proj_b, x, x2);

    ln2_kernel<<<MROWS / 8, 256, 0, stream>>>(x2, ln2_w, ln2_b, xn);

    fc1_gemm_kernel<<<MROWS / 16 / 8, 256, 0, stream>>>(xn, wfc1, fc1_b, hbuf);

    fc2_gemm_kernel<<<MROWS / 16 / 8, 256, 0, stream>>>(hbuf, wfc2, fc2_b, x2, out);
}
